// TK_79955111182554
// MI455X (gfx1250) — compile-verified, matrix operand order checked by reading, not run on hardware
//
#include <hip/hip_runtime.h>
#include <hip/hip_bf16.h>

// ---------------------------------------------------------------------------
// KNRM-style ranker on gfx1250 (CDNA5, wave32, WMMA + TDM).
// Linear layers + attention matmuls -> v_wmma_f32_16x16x32_f16.
// GEMM weight tiles staged to LDS once per 8-wave block via the Tensor Data
// Mover (tensor_load_to_lds / s_wait_tensorcnt). Numerically sensitive parts
// (softmax stats, LayerNorm, cosine + RBF pooling with sigma=1e-4) are f32.
// ---------------------------------------------------------------------------

typedef __attribute__((ext_vector_type(16))) _Float16 v16h;
typedef __attribute__((ext_vector_type(8)))  _Float16 v8h;
typedef __attribute__((ext_vector_type(8)))  float    v8f;
typedef __attribute__((ext_vector_type(4)))  unsigned int v4u;
typedef __attribute__((ext_vector_type(4)))  int      v4i;
typedef __attribute__((ext_vector_type(8)))  int      v8i;

#if defined(__has_builtin)
#if __has_builtin(__builtin_amdgcn_tensor_load_to_lds) && \
    __has_builtin(__builtin_amdgcn_s_wait_tensorcnt)
#define HAVE_TDM 1
#endif
#endif
#ifndef HAVE_TDM
#define HAVE_TDM 0
#endif

namespace cfg {
constexpr int B   = 256;
constexpr int Q   = 32;
constexpr int D   = 256;
constexpr int E   = 300;
constexpr int H   = 10;
constexpr int DH  = 30;   // E / H
constexpr int DFF = 100;
constexpr int NK  = 11;
constexpr int KE  = 320;  // E padded to 32
constexpr int KF  = 128;  // DFF padded to 32
}

union FragH { v16h v; v8h h[2]; };
union AccF  { v8f  v; float f[8]; };

// Load one 16x32 f16 A/B fragment row-slice for this lane, following the
// documented CDNA5 16-bit operand K-striping:
//   lanes 0-15 : VGPR0-3 = K 0..7,  VGPR4-7 = K 16..23
//   lanes 16-31: VGPR0-3 = K 8..15, VGPR4-7 = K 24..31
__device__ __forceinline__ v16h load_frag(const _Float16* row, int kb, int half) {
  const _Float16* p = row + kb * 32 + half * 8;
  FragH f;
  f.h[0] = *(const v8h*)(p);        // K base .. base+7
  f.h[1] = *(const v8h*)(p + 16);   // K base+16 .. base+23
  return f.v;
}

__device__ __forceinline__ v8f wmma_f16(v16h a, v16h b, v8f c) {
  return __builtin_amdgcn_wmma_f32_16x16x32_f16(false, a, false, b,
                                                (short)0, c, false, false);
}

// ---------------------------------------------------------------------------
// x[b,s,:] = emb[tok[b,s]] + sinusoidal positional encoding
// ---------------------------------------------------------------------------
__global__ void embed_pe(const int* __restrict__ toks,
                         const float* __restrict__ emb,
                         float* __restrict__ x, int S) {
  const long t = blockIdx.x;
  const int  s = (int)(t % S);
  const int  tok = toks[t];
  const float* erow = emb + (long)tok * cfg::E;
  float* xrow = x + t * cfg::E;
  for (int e = threadIdx.x; e < cfg::E; e += blockDim.x) {
    const int i2 = e & ~1;
    const float div = __expf(-(float)i2 * (9.210340371976184f / (float)cfg::E));
    const float ang = (float)s * div;
    const float pe  = (e & 1) ? __cosf(ang) : __sinf(ang);
    xrow[e] = erow[e] + pe;
  }
}

// ---------------------------------------------------------------------------
// f32 (R x C) -> zero-padded f16 (gridDim.x x Cp)
// ---------------------------------------------------------------------------
__global__ void convert_pad(const float* __restrict__ src,
                            _Float16* __restrict__ dst,
                            int R, int C, int Cp) {
  const long r = blockIdx.x;
  for (int c = threadIdx.x; c < Cp; c += blockDim.x) {
    float v = (r < R && c < C) ? src[r * C + c] : 0.0f;
    dst[r * Cp + c] = (_Float16)v;
  }
}

// ---------------------------------------------------------------------------
// out[M,N] = A[M,Kp] * W[N,Kp]^T + bias  (optional ReLU)
// 8 waves per block; all waves share one 16-row weight tile staged into LDS
// by the Tensor Data Mover. Wave w computes M-tile (blockIdx.x*8+w).
// ---------------------------------------------------------------------------
__global__ void __launch_bounds__(256)
gemm_wmma(const _Float16* __restrict__ A, const _Float16* __restrict__ W,
          const float* __restrict__ bias, float* __restrict__ out,
          int N, int Kp, int relu) {
  __shared__ __align__(16) _Float16 wt[16 * cfg::KE];  // packed stride = Kp

  const int tid  = threadIdx.x;
  const int lane = tid & 31;
  const int wave = tid >> 5;
  const int half = lane >> 4;
  const int li   = lane & 15;
  const long m0  = ((long)blockIdx.x * 8 + wave) * 16;
  const int  n0  = blockIdx.y * 16;

#if HAVE_TDM
  // ---- stage W[n0 .. n0+15][0 .. Kp) into LDS via TDM (wave 0 only) -------
  if (wave == 0) {
    const unsigned long long ga =
        (unsigned long long)(const void*)(W + (long)n0 * Kp);
    const unsigned int la = (unsigned int)(unsigned long long)(const void*)&wt[0];
    // D# group 0: count=1 | lds_addr | global_addr[56:0] | type=2
    v4u g0 = { 1u, la, (unsigned int)ga,
               (unsigned int)(((ga >> 32) & 0x01FFFFFFull) | 0x80000000u) };
    // D# group 1: data_size=1 (2B) ; tensor_dim0=Kp ; tensor_dim1=16 ;
    //             tile_dim0=Kp ; tile_dim1=16 ; tensor_dim0_stride=Kp
    v8i g1 = { (int)(1u << 16),             // workgroup_mask=0, data_size=2B
               (int)((unsigned)Kp << 16),   // tensor_dim0[15:0] in bits 63:48
               (int)(16u << 16),            // tensor_dim0 hi=0, tensor_dim1=16
               (int)((unsigned)Kp << 16),   // tensor_dim1 hi=0, tile_dim0=Kp
               16,                          // tile_dim1=16, tile_dim2=0
               Kp,                          // tensor_dim0_stride low 32
               0, 0 };                      // stride hi / dim1_stride
    v4i g2 = { 1, 0, 0, 0 };               // tensor_dim2=1 (2D tile)
    v4i g3 = { 0, 0, 0, 0 };
#if __clang_major__ >= 23
    v8i g4 = { 0, 0, 0, 0, 0, 0, 0, 0 };
    __builtin_amdgcn_tensor_load_to_lds(g0, g1, g2, g3, g4, 0);
#else
    __builtin_amdgcn_tensor_load_to_lds(g0, g1, g2, g3, 0);
#endif
    __builtin_amdgcn_s_wait_tensorcnt((unsigned short)0);
  }
#else
  for (int idx = tid; idx < 16 * Kp; idx += 256) {
    const int r = idx / Kp, c = idx % Kp;
    wt[r * Kp + c] = W[(long)(n0 + r) * Kp + c];
  }
#endif
  __syncthreads();

  const _Float16* arow = A + (m0 + li) * (long)Kp;
  const _Float16* brow = &wt[li * Kp];

  AccF acc;
#pragma unroll
  for (int i = 0; i < 8; ++i) acc.f[i] = 0.0f;

  const int ksteps = Kp >> 5;
  // Software pipeline: keep next A fragment's loads in flight across the WMMA.
  v16h a_cur = load_frag(arow, 0, half);
  for (int kb = 0; kb < ksteps; ++kb) {
    const int kn = (kb + 1 < ksteps) ? kb + 1 : kb;
    v16h a_nxt = load_frag(arow, kn, half);
    if (kb + 2 < ksteps)
      __builtin_prefetch(arow + (kb + 2) * 32, 0, 3);   // global_prefetch_b8
    v16h b = load_frag(brow, kb, half);                 // ds_load_b128
    acc.v = wmma_f16(a_cur, b, acc.v);
    a_cur = a_nxt;
  }

  const int n = n0 + li;
  if (n < N) {
    const float bv = bias[n];
#pragma unroll
    for (int v = 0; v < 8; ++v) {
      const long m = m0 + v + half * 8;      // documented C/D layout
      float r = acc.f[v] + bv;
      if (relu) r = fmaxf(r, 0.0f);
      out[m * N + n] = r;
    }
  }
}

// ---------------------------------------------------------------------------
// Split fused QKV (tokens x 900) into padded per-head f16 buffers:
//   qh,kh : (B,H,S,32)   (dh padded 30->32, q pre-scaled by 1/sqrt(30))
//   vth   : (B,H,32,S)   (V transposed so PV B-fragments are K-contiguous)
// ---------------------------------------------------------------------------
__global__ void qkv_split(const float* __restrict__ qkv,
                          _Float16* __restrict__ qh,
                          _Float16* __restrict__ kh,
                          _Float16* __restrict__ vth, int S) {
  const long t = blockIdx.x;
  const int b = (int)(t / S), s = (int)(t % S);
  const float* row = qkv + t * (3 * cfg::E);
  for (int idx = threadIdx.x; idx < cfg::H * 32; idx += blockDim.x) {
    const int h = idx >> 5, d = idx & 31;
    float qv = 0.f, kv = 0.f, vv = 0.f;
    if (d < cfg::DH) {
      qv = row[h * cfg::DH + d];
      kv = row[cfg::E + h * cfg::DH + d];
      vv = row[2 * cfg::E + h * cfg::DH + d];
    }
    const long bh   = (long)b * cfg::H + h;
    const long base = (bh * S + s) * 32 + d;
    qh[base] = (_Float16)(qv * 0.18257418583505537f);  // 1/sqrt(30)
    kh[base] = (_Float16)kv;
    vth[(bh * 32 + d) * S + s] = (_Float16)vv;
  }
}

// ---------------------------------------------------------------------------
// Attention: one wave per (b*H+h, 16-query tile).
//   scores = q k^T (WMMA, K=32), f32 softmax in LDS,
//   probs requantized to f16 in LDS, out = P V (WMMA over K=S).
// ---------------------------------------------------------------------------
__global__ void __launch_bounds__(32)
attn_wmma(const _Float16* __restrict__ qh, const _Float16* __restrict__ kh,
          const _Float16* __restrict__ vth, float* __restrict__ o, int S) {
  __shared__ __align__(16) float    sc[16][256];
  __shared__ __align__(16) _Float16 ph[16][256];
  __shared__ float rsum[16];

  const int lane = threadIdx.x & 31;
  const int half = lane >> 4;
  const int li   = lane & 15;
  const int bh   = blockIdx.x;
  const int b    = bh / cfg::H, h = bh % cfg::H;
  const int q0   = blockIdx.y * 16;

  const _Float16* qbase = qh  + (long)bh * S * 32;
  const _Float16* kbase = kh  + (long)bh * S * 32;
  const _Float16* vbase = vth + (long)bh * 32 * S;

  // --- scores = q k^T ------------------------------------------------------
  const v16h aq = load_frag(qbase + (long)(q0 + li) * 32, 0, half);
  for (int nt = 0; nt < S / 16; ++nt) {
    AccF acc;
#pragma unroll
    for (int i = 0; i < 8; ++i) acc.f[i] = 0.0f;
    v16h bk = load_frag(kbase + (long)(nt * 16 + li) * 32, 0, half);
    acc.v = wmma_f16(aq, bk, acc.v);
#pragma unroll
    for (int v = 0; v < 8; ++v)
      sc[v + half * 8][nt * 16 + li] = acc.f[v];
  }
  __syncthreads();

  // --- row softmax (f32), store probs as f16 for A fragments ---------------
  if (lane < 16) {
    float mx = -1e30f;
    for (int j = 0; j < S; ++j) mx = fmaxf(mx, sc[lane][j]);
    float sum = 0.0f;
    for (int j = 0; j < S; ++j) {
      const float e = __expf(sc[lane][j] - mx);
      ph[lane][j] = (_Float16)e;
      sum += e;
    }
    rsum[lane] = sum;
  }
  __syncthreads();

  // --- out = P V -----------------------------------------------------------
  for (int nt = 0; nt < 2; ++nt) {          // dh 30 -> two 16-wide N tiles
    AccF acc;
#pragma unroll
    for (int i = 0; i < 8; ++i) acc.f[i] = 0.0f;
    for (int kb = 0; kb < S / 32; ++kb) {
      v16h ap = load_frag(&ph[li][0], kb, half);
      v16h bv = load_frag(vbase + (long)(nt * 16 + li) * S, kb, half);
      acc.v = wmma_f16(ap, bv, acc.v);
    }
    const int n = nt * 16 + li;
    if (n < cfg::DH) {
#pragma unroll
      for (int v = 0; v < 8; ++v) {
        const int m = v + half * 8;
        const float val = acc.f[v] / rsum[m];
        o[((long)b * S + q0 + m) * cfg::E + h * cfg::DH + n] = val;
      }
    }
  }
}

// ---------------------------------------------------------------------------
// x = LayerNorm(x + y) * g + b    (one wave32 per token, shfl reductions)
// ---------------------------------------------------------------------------
__global__ void __launch_bounds__(32)
residual_ln(float* __restrict__ x, const float* __restrict__ y,
            const float* __restrict__ g, const float* __restrict__ bt) {
  const long t = blockIdx.x;
  const int lane = threadIdx.x;
  float v[10];
  int cnt = 0;
  float sum = 0.0f;
  for (int e = lane; e < cfg::E; e += 32) {
    const float a = x[t * cfg::E + e] + y[t * cfg::E + e];
    v[cnt++] = a;
    sum += a;
  }
#pragma unroll
  for (int o = 16; o > 0; o >>= 1) sum += __shfl_xor(sum, o, 32);
  const float mean = sum * (1.0f / cfg::E);
  float sq = 0.0f;
  for (int i = 0; i < cnt; ++i) { const float d = v[i] - mean; sq += d * d; }
#pragma unroll
  for (int o = 16; o > 0; o >>= 1) sq += __shfl_xor(sq, o, 32);
  const float inv = rsqrtf(sq * (1.0f / cfg::E) + 1e-5f);
  cnt = 0;
  for (int e = lane; e < cfg::E; e += 32)
    x[t * cfg::E + e] = (v[cnt++] - mean) * inv * g[e] + bt[e];
}

// ---------------------------------------------------------------------------
// Final scoring: per batch b, cosine translation matrix + RBF kernel pooling.
// f32 throughout (sigma0 = 1e-4 is far below f16 resolution near tm=1).
// ---------------------------------------------------------------------------
__global__ void __launch_bounds__(256)
score_kernel(const float* __restrict__ qe, const float* __restrict__ de,
             const int* __restrict__ qt, const int* __restrict__ dt,
             const float* __restrict__ fcw, float* __restrict__ out) {
  __shared__ float qn[32][301];
  __shared__ float dch[16][301];
  __shared__ float kacc[32][cfg::NK];
  __shared__ float qm[32], dm[16];
  __shared__ float res;

  const int b = blockIdx.x, tid = threadIdx.x;

  for (int idx = tid; idx < 32 * cfg::E; idx += 256) {
    const int q = idx / cfg::E, e = idx % cfg::E;
    qn[q][e] = qe[((long)b * cfg::Q + q) * cfg::E + e];
  }
  for (int idx = tid; idx < 32 * cfg::NK; idx += 256)
    kacc[idx / cfg::NK][idx % cfg::NK] = 0.0f;
  if (tid == 0) res = 0.0f;
  __syncthreads();

  if (tid < 32) {
    float ss = 0.0f;
    for (int e = 0; e < cfg::E; ++e) { const float a = qn[tid][e]; ss += a * a; }
    const float rn = rsqrtf(ss);
    for (int e = 0; e < cfg::E; ++e) qn[tid][e] *= rn;
    qm[tid] = (qt[b * cfg::Q + tid] > 0) ? 1.0f : 0.0f;
  }
  __syncthreads();

  const float mu[cfg::NK]  = {1.0f, 0.9f, 0.7f, 0.5f, 0.3f, 0.1f,
                              -0.1f, -0.3f, -0.5f, -0.7f, -0.9f};
  const float i2s[cfg::NK] = {5.0e7f, 50.f, 50.f, 50.f, 50.f, 50.f,
                              50.f, 50.f, 50.f, 50.f, 50.f};  // 1/(2*sigma^2)

  for (int ch = 0; ch < cfg::D / 16; ++ch) {
    for (int idx = tid; idx < 16 * cfg::E; idx += 256) {
      const int d = idx / cfg::E, e = idx % cfg::E;
      dch[d][e] = de[((long)b * cfg::D + ch * 16 + d) * cfg::E + e];
    }
    __syncthreads();
    if (tid < 16) {
      float ss = 0.0f;
      for (int e = 0; e < cfg::E; ++e) { const float a = dch[tid][e]; ss += a * a; }
      const float rn = rsqrtf(ss);
      for (int e = 0; e < cfg::E; ++e) dch[tid][e] *= rn;
      dm[tid] = (dt[b * cfg::D + ch * 16 + tid] > 0) ? 1.0f : 0.0f;
    }
    __syncthreads();
    for (int p = tid; p < 32 * 16; p += 256) {
      const int q = p >> 4, d = p & 15;
      if (qm[q] * dm[d] > 0.0f) {
        float dot = 0.0f;
        for (int e = 0; e < cfg::E; ++e) dot += qn[q][e] * dch[d][e];
#pragma unroll
        for (int k = 0; k < cfg::NK; ++k) {
          const float df = dot - mu[k];
          atomicAdd(&kacc[q][k], __expf(-df * df * i2s[k]));  // ds_add_f32
        }
      }
    }
    __syncthreads();
  }

  if (tid < cfg::NK) {
    float f = 0.0f;
    for (int q = 0; q < cfg::Q; ++q)
      f += __logf(fmaxf(kacc[q][tid], 1e-10f));
    atomicAdd(&res, f * fcw[tid]);
  }
  __syncthreads();
  if (tid == 0) out[b] = res;
}

// ---------------------------------------------------------------------------
// Host-side orchestration
// ---------------------------------------------------------------------------
extern "C" void kernel_launch(void* const* d_in, const int* in_sizes, int n_in,
                              void* d_out, int out_size, void* d_ws, size_t ws_size,
                              hipStream_t stream) {
  (void)in_sizes; (void)n_in; (void)out_size; (void)ws_size;
  using namespace cfg;

  const float* emb  = (const float*)d_in[0];
  const float* Wqkv = (const float*)d_in[1];
  const float* bqkv = (const float*)d_in[2];
  const float* Wo   = (const float*)d_in[3];
  const float* bo   = (const float*)d_in[4];
  const float* ln1g = (const float*)d_in[5];
  const float* ln1b = (const float*)d_in[6];
  const float* W1   = (const float*)d_in[7];
  const float* b1   = (const float*)d_in[8];
  const float* W2   = (const float*)d_in[9];
  const float* b2   = (const float*)d_in[10];
  const float* ln2g = (const float*)d_in[11];
  const float* ln2b = (const float*)d_in[12];
  const float* fcw  = (const float*)d_in[13];
  const int*   qtok = (const int*)d_in[14];
  const int*   dtok = (const int*)d_in[15];

  char* base = (char*)d_ws;
  size_t off = 0;
  auto alloc = [&](size_t bytes) -> void* {
    void* p = base + off;
    off += (bytes + 255) & ~(size_t)255;
    return p;
  };

  const long BtokD = (long)B * D;            // 65536 (max token count)
  float*    x_q   = (float*)alloc((size_t)B * Q * E * 4);
  float*    x_d   = (float*)alloc((size_t)BtokD * E * 4);
  _Float16* xh    = (_Float16*)alloc((size_t)BtokD * KE * 2);
  float*    qkv   = (float*)alloc((size_t)BtokD * 3 * E * 4);
  _Float16* qh    = (_Float16*)alloc((size_t)B * H * D * 32 * 2);
  _Float16* kh    = (_Float16*)alloc((size_t)B * H * D * 32 * 2);
  _Float16* vth   = (_Float16*)alloc((size_t)B * H * 32 * D * 2);
  float*    obuf  = (float*)alloc((size_t)BtokD * E * 4);
  float*    ybuf  = (float*)alloc((size_t)BtokD * E * 4);
  float*    h1    = (float*)alloc((size_t)BtokD * DFF * 4);
  _Float16* h1h   = (_Float16*)alloc((size_t)BtokD * KF * 2);
  _Float16* wqkvh = (_Float16*)alloc((size_t)912 * KE * 2);   // 900 -> 912 rows
  _Float16* woh   = (_Float16*)alloc((size_t)304 * KE * 2);   // 300 -> 304
  _Float16* w1h   = (_Float16*)alloc((size_t)112 * KE * 2);   // 100 -> 112
  _Float16* w2h   = (_Float16*)alloc((size_t)304 * KF * 2);

  for (int pass = 0; pass < 2; ++pass) {
    const int  S    = (pass == 0) ? Q : D;
    const long Btok = (long)B * S;
    float*     x    = (pass == 0) ? x_q : x_d;
    const int* toks = (pass == 0) ? qtok : dtok;
    const unsigned mblk = (unsigned)(Btok / 128);   // 8 M-tiles per block

    embed_pe<<<dim3((unsigned)Btok), dim3(128), 0, stream>>>(toks, emb, x, S);

    for (int l = 0; l < 2; ++l) {
      convert_pad<<<dim3(912), dim3(128), 0, stream>>>(
          Wqkv + (size_t)l * 3 * E * E, wqkvh, 3 * E, E, KE);
      convert_pad<<<dim3(304), dim3(128), 0, stream>>>(
          Wo + (size_t)l * E * E, woh, E, E, KE);
      convert_pad<<<dim3(112), dim3(128), 0, stream>>>(
          W1 + (size_t)l * DFF * E, w1h, DFF, E, KE);
      convert_pad<<<dim3(304), dim3(128), 0, stream>>>(
          W2 + (size_t)l * E * DFF, w2h, E, DFF, KF);

      // --- self-attention block ---
      convert_pad<<<dim3((unsigned)Btok), dim3(128), 0, stream>>>(
          x, xh, (int)Btok, E, KE);
      gemm_wmma<<<dim3(mblk, 57), dim3(256), 0, stream>>>(
          xh, wqkvh, bqkv + l * 3 * E, qkv, 3 * E, KE, 0);
      qkv_split<<<dim3((unsigned)Btok), dim3(128), 0, stream>>>(
          qkv, qh, kh, vth, S);
      attn_wmma<<<dim3(B * H, S / 16), dim3(32), 0, stream>>>(
          qh, kh, vth, obuf, S);
      convert_pad<<<dim3((unsigned)Btok), dim3(128), 0, stream>>>(
          obuf, xh, (int)Btok, E, KE);
      gemm_wmma<<<dim3(mblk, 19), dim3(256), 0, stream>>>(
          xh, woh, bo + l * E, ybuf, E, KE, 0);
      residual_ln<<<dim3((unsigned)Btok), dim3(32), 0, stream>>>(
          x, ybuf, ln1g + l * E, ln1b + l * E);

      // --- feed-forward block ---
      convert_pad<<<dim3((unsigned)Btok), dim3(128), 0, stream>>>(
          x, xh, (int)Btok, E, KE);
      gemm_wmma<<<dim3(mblk, 7), dim3(256), 0, stream>>>(
          xh, w1h, b1 + l * DFF, h1, DFF, KE, 1);
      convert_pad<<<dim3((unsigned)Btok), dim3(128), 0, stream>>>(
          h1, h1h, (int)Btok, DFF, KF);
      gemm_wmma<<<dim3(mblk, 19), dim3(256), 0, stream>>>(
          h1h, w2h, b2 + l * E, ybuf, E, KF, 0);
      residual_ln<<<dim3((unsigned)Btok), dim3(32), 0, stream>>>(
          x, ybuf, ln2g + l * E, ln2b + l * E);
    }
  }

  score_kernel<<<dim3(B), dim3(256), 0, stream>>>(
      x_q, x_d, qtok, dtok, fcw, (float*)d_out);
}
